// BertParser_22050362098343
// MI455X (gfx1250) — compile-verified, matrix operand order checked by reading, not run on hardware
//
#include <hip/hip_runtime.h>
#include <hip/hip_bf16.h>
#include <math.h>

// Problem sizes (fixed by the reference)
#define N_TOK 4096
#define N_ARC 262144
#define HDIM  768
#define D_HID 512
#define N_OUT (2 * D_HID)   // concatenated [Ph | Pd] projection width = 1024

typedef _Float16 v8h  __attribute__((ext_vector_type(8)));
typedef _Float16 v16h __attribute__((ext_vector_type(16)));
typedef float    v8f  __attribute__((ext_vector_type(8)));

// Native CDNA5 V_TANH_F32 when available (TRANS op), ocml fallback otherwise.
static __device__ __forceinline__ float fast_tanh(float x) {
#if __has_builtin(__builtin_amdgcn_tanhf)
    return __builtin_amdgcn_tanhf(x);
#else
    return tanhf(x);
#endif
}

// ---------------------------------------------------------------------------
// Kernel 0a: f32 -> f16 conversion of embeddings (4096 x 768)
// ---------------------------------------------------------------------------
__global__ void cvt_embed(const float* __restrict__ E, _Float16* __restrict__ Eh, int n) {
    int i = blockIdx.x * blockDim.x + threadIdx.x;
    if (i < n) Eh[i] = (_Float16)E[i];
}

// ---------------------------------------------------------------------------
// Kernel 0b: build Bh (1024 x 768, f16) from W_hidden (512 x 1536, f32).
// Row j < 512  -> Wh row j     = W_hidden[j][0:768]
// Row j >= 512 -> Wd row j-512 = W_hidden[j-512][768:1536]
// So P = E @ Bh^T gives P[t] = [Ph[t] | Pd[t]].
// ---------------------------------------------------------------------------
__global__ void build_w(const float* __restrict__ W, _Float16* __restrict__ Bh) {
    int i = blockIdx.x * blockDim.x + threadIdx.x;
    if (i < N_OUT * HDIM) {
        int j = i / HDIM;
        int h = i - j * HDIM;
        int d  = j & (D_HID - 1);          // j % 512
        int ko = (j >> 9) * HDIM;          // 0 or 768
        Bh[i] = (_Float16)W[d * (2 * HDIM) + ko + h];
    }
}

// ---------------------------------------------------------------------------
// Fragment loader for one k-step: 2xA + 8xB global_load_b128.
// ---------------------------------------------------------------------------
static __device__ __forceinline__ void load_frags(
        const _Float16* __restrict__ ar, const _Float16* __restrict__ br, int kk,
        v16h& A, v16h& B0, v16h& B1, v16h& B2, v16h& B3) {
    v8h a0  = *(const v8h*)(ar + kk);
    v8h a1  = *(const v8h*)(ar + kk + 16);
    v8h b00 = *(const v8h*)(br + 0 * 16 * HDIM + kk);
    v8h b01 = *(const v8h*)(br + 0 * 16 * HDIM + kk + 16);
    v8h b10 = *(const v8h*)(br + 1 * 16 * HDIM + kk);
    v8h b11 = *(const v8h*)(br + 1 * 16 * HDIM + kk + 16);
    v8h b20 = *(const v8h*)(br + 2 * 16 * HDIM + kk);
    v8h b21 = *(const v8h*)(br + 2 * 16 * HDIM + kk + 16);
    v8h b30 = *(const v8h*)(br + 3 * 16 * HDIM + kk);
    v8h b31 = *(const v8h*)(br + 3 * 16 * HDIM + kk + 16);
    A  = __builtin_shufflevector(a0,  a1,  0,1,2,3,4,5,6,7,8,9,10,11,12,13,14,15);
    B0 = __builtin_shufflevector(b00, b01, 0,1,2,3,4,5,6,7,8,9,10,11,12,13,14,15);
    B1 = __builtin_shufflevector(b10, b11, 0,1,2,3,4,5,6,7,8,9,10,11,12,13,14,15);
    B2 = __builtin_shufflevector(b20, b21, 0,1,2,3,4,5,6,7,8,9,10,11,12,13,14,15);
    B3 = __builtin_shufflevector(b30, b31, 0,1,2,3,4,5,6,7,8,9,10,11,12,13,14,15);
}

// ---------------------------------------------------------------------------
// Kernel 1: WMMA GEMM  P(4096x1024, f32) = Eh(4096x768) @ Bh(1024x768)^T
// Each wave owns a 16x64 strip: ONE A fragment per k-step feeds FOUR
// independent accumulators. Explicit 1-deep software pipeline: k-step i+1's
// fragments are loaded into a SECOND set of live registers before the four
// WMMAs of k-step i issue, so the pre-WMMA wait is partial instead of a
// full loadcnt drain. __launch_bounds__(256, 1) lifts the default
// occupancy-driven VGPR cap (~120) so both pipeline stages (2x40 fragment
// VGPRs + 32 accumulator VGPRs) stay in registers with NO scratch spills;
// actual occupancy is still governed by real usage (~6 waves/SIMD).
// Fragment addressing follows the CDNA5 16-bit A-layout:
//   lane L: row = L&15, half = L>>4
//   VGPRs 0-3 hold K = half*8 + 0..7, VGPRs 4-7 hold K = 16 + half*8 + 0..7
// B^T column n is a contiguous row of Bh -> identical addressing pattern.
// ---------------------------------------------------------------------------
__global__ void __launch_bounds__(256, 1)
proj_gemm(const _Float16* __restrict__ Eh,
          const _Float16* __restrict__ Bh,
          float* __restrict__ P) {
    const int lane = threadIdx.x & 31;
    const int wave = threadIdx.x >> 5;
    const int task = blockIdx.x * 8 + wave;          // 4096 strip tasks
    const int mtile = task >> 4;                     // 16 N-groups of 64
    const int ngrp  = task & 15;

    const int r    = lane & 15;                      // A row / B^T column
    const int half = lane >> 4;
    const int kb   = half * 8;

    const _Float16* ar = Eh + (size_t)(mtile * 16 + r) * HDIM + kb;
    const _Float16* br = Bh + (size_t)(ngrp * 64 + r) * HDIM + kb;

    v8f acc0 = {}, acc1 = {}, acc2 = {}, acc3 = {};

    // Prologue: fragments for k-step 0
    v16h A, B0, B1, B2, B3;
    load_frags(ar, br, 0, A, B0, B1, B2, B3);

    for (int kk = 32; kk < HDIM; kk += 32) {
        // Prefetch next k-step into a distinct live set (stays in flight
        // across the WMMAs below).
        v16h An, B0n, B1n, B2n, B3n;
        load_frags(ar, br, kk, An, B0n, B1n, B2n, B3n);

        // 4 independent WMMA chains on the current set
        acc0 = __builtin_amdgcn_wmma_f32_16x16x32_f16(false, A, false, B0, (short)0, acc0, false, false);
        acc1 = __builtin_amdgcn_wmma_f32_16x16x32_f16(false, A, false, B1, (short)0, acc1, false, false);
        acc2 = __builtin_amdgcn_wmma_f32_16x16x32_f16(false, A, false, B2, (short)0, acc2, false, false);
        acc3 = __builtin_amdgcn_wmma_f32_16x16x32_f16(false, A, false, B3, (short)0, acc3, false, false);

        A = An; B0 = B0n; B1 = B1n; B2 = B2n; B3 = B3n;
    }

    // Epilogue: last k-step
    acc0 = __builtin_amdgcn_wmma_f32_16x16x32_f16(false, A, false, B0, (short)0, acc0, false, false);
    acc1 = __builtin_amdgcn_wmma_f32_16x16x32_f16(false, A, false, B1, (short)0, acc1, false, false);
    acc2 = __builtin_amdgcn_wmma_f32_16x16x32_f16(false, A, false, B2, (short)0, acc2, false, false);
    acc3 = __builtin_amdgcn_wmma_f32_16x16x32_f16(false, A, false, B3, (short)0, acc3, false, false);

    // C/D layout: VGPR i -> row (half*8 + i), lane&15 -> column
    float* pbase = P + (size_t)(mtile * 16 + half * 8) * N_OUT + ngrp * 64 + r;
    #pragma unroll
    for (int i = 0; i < 8; ++i) {
        pbase[(size_t)i * N_OUT +  0] = acc0[i];
        pbase[(size_t)i * N_OUT + 16] = acc1[i];
        pbase[(size_t)i * N_OUT + 32] = acc2[i];
        pbase[(size_t)i * N_OUT + 48] = acc3[i];
    }
}

// ---------------------------------------------------------------------------
// Kernel 2: arc scoring. One wave (32 lanes) per arc; each lane covers 16 of
// the 512 hidden dims as 4x float4 loads (all gather traffic is L2-resident:
// P is 16 MB << 192 MB L2). tanh uses the native CDNA5 TRANS op.
// ---------------------------------------------------------------------------
__global__ void score_arcs(const float* __restrict__ P,
                           const float* __restrict__ bhid,   // 512
                           const float* __restrict__ Wout,   // 512
                           const float* __restrict__ bout,   // 1
                           const int*   __restrict__ head,
                           const int*   __restrict__ dep,
                           float* __restrict__ out) {
    const int lane = threadIdx.x & 31;
    const int wave = threadIdx.x >> 5;
    const int arc  = blockIdx.x * 8 + wave;

    const float* ph = P + (size_t)head[arc] * N_OUT;          // Ph row
    const float* pd = P + (size_t)dep[arc]  * N_OUT + D_HID;  // Pd row
    const int d0 = lane * 16;

    float s = 0.0f;
    #pragma unroll
    for (int c = 0; c < 4; ++c) {
        const int d = d0 + c * 4;
        float4 a = *(const float4*)(ph   + d);
        float4 b = *(const float4*)(pd   + d);
        float4 g = *(const float4*)(bhid + d);
        float4 w = *(const float4*)(Wout + d);
        s += w.x * fast_tanh(a.x + b.x + g.x);
        s += w.y * fast_tanh(a.y + b.y + g.y);
        s += w.z * fast_tanh(a.z + b.z + g.z);
        s += w.w * fast_tanh(a.w + b.w + g.w);
    }
    // wave32 butterfly reduction
    #pragma unroll
    for (int off = 16; off > 0; off >>= 1) s += __shfl_xor(s, off, 32);
    if (lane == 0) out[arc] = s + bout[0];
}

// ---------------------------------------------------------------------------
extern "C" void kernel_launch(void* const* d_in, const int* in_sizes, int n_in,
                              void* d_out, int out_size, void* d_ws, size_t ws_size,
                              hipStream_t stream) {
    (void)in_sizes; (void)n_in; (void)out_size; (void)ws_size;
    const float* E    = (const float*)d_in[0];   // 4096 x 768
    const float* W    = (const float*)d_in[1];   // 512 x 1536
    const float* bhid = (const float*)d_in[2];   // 512
    const float* Wout = (const float*)d_in[3];   // 1 x 512
    const float* bout = (const float*)d_in[4];   // 1
    const int*   head = (const int*)d_in[5];     // 262144
    const int*   dep  = (const int*)d_in[6];     // 262144
    float* out = (float*)d_out;

    // Workspace layout (all 16B-aligned offsets):
    //   P  : 4096*1024 f32 = 16 MB   @ 0
    //   Eh : 4096*768  f16 =  6 MB   @ 16 MB
    //   Bh : 1024*768  f16 = 1.5 MB  @ 22 MB
    char* ws = (char*)d_ws;
    float*    P  = (float*)ws;
    _Float16* Eh = (_Float16*)(ws + ((size_t)16 << 20));
    _Float16* Bh = (_Float16*)(ws + ((size_t)22 << 20));

    const int nE = N_TOK * HDIM;
    const int nB = N_OUT * HDIM;
    cvt_embed<<<(nE + 255) / 256, 256, 0, stream>>>(E, Eh, nE);
    build_w  <<<(nB + 255) / 256, 256, 0, stream>>>(W, Bh);

    // 4096 strip tasks (256 M-tiles x 16 N-groups of 64), 8 waves per block
    proj_gemm<<<4096 / 8, 256, 0, stream>>>(Eh, Bh, P);

    // 262144 arcs, one wave each, 8 waves per block
    score_arcs<<<N_ARC / 8, 256, 0, stream>>>(P, bhid, Wout, bout, head, dep, out);
}
